// RNN_1TM_76544907149440
// MI455X (gfx1250) — matrix operand order repaired and checked by reading, not compiled, not run on hardware
//
#include <hip/hip_runtime.h>
#include <math.h>

typedef __bf16 bf16_t;
typedef __attribute__((ext_vector_type(16))) __bf16 v16bf;
typedef __attribute__((ext_vector_type(8)))  __bf16 v8bf;
typedef __attribute__((ext_vector_type(8)))  float  v8f;

#define BATCH 128
#define IN_DIM 256
#define HID    512
#define SEQ    1024
#define NWG    32      // 32 WGs * 16 hidden cols = 512
#define NTHR   256     // 8 wave32 per WG; wave w owns batch rows 16w..16w+15

__device__ __forceinline__ float fast_sigmoid(float x) {
    return 1.0f / (1.0f + __expf(-x));
}
__device__ __forceinline__ float fast_tanh(float x) {
    float e = __expf(-2.0f * x);
    return (1.0f - e) / (1.0f + e);
}

// Grid-wide barrier: monotonically increasing counter, one signal per WG.
__device__ __forceinline__ void grid_barrier(unsigned* counter, unsigned target) {
    __threadfence();          // release all prior global stores
    __syncthreads();          // whole block arrived (stores issued)
    if (threadIdx.x == 0) {
        atomicAdd(counter, 1u);
        while (atomicAdd(counter, 0u) < target) {
            __builtin_amdgcn_s_sleep(1);
        }
    }
    __syncthreads();
    __threadfence();          // acquire: don't read stale h
}

__global__ __launch_bounds__(NTHR)
void gru_decoder_persistent(const float* __restrict__ x,      // [128,256]
                            const float* __restrict__ w_ih,   // [1536,256]
                            const float* __restrict__ w_hh,   // [1536,512]
                            const float* __restrict__ b_ih,   // [1536]
                            const float* __restrict__ b_hh,   // [1536]
                            float* __restrict__ out,          // [128,1024,512]
                            unsigned* __restrict__ counter,
                            bf16_t* __restrict__ hbuf)        // 2 x [128,512] bf16
{
    // LDS: this WG's slice of w_hh in bf16: [gate(3)][col(16)][k(512)]
    __shared__ __align__(32) bf16_t wlds[3 * 16 * HID];   // 49152 bytes

    const int tid  = threadIdx.x;
    const int wg   = blockIdx.x;        // 0..31
    const int lane = tid & 31;
    const int wave = tid >> 5;          // 0..7
    const int hi   = lane >> 4;         // half-wave select
    const int ln   = lane & 15;         // N within tile (C layout), M for A loads
    const int j    = wg * 16;           // hidden column base for this WG
    const int rowbase = wave * 16;      // batch row base for this wave

    // ---- One-time: load + convert w_hh slice to bf16 in LDS ----
    for (int idx = tid; idx < 3 * 16 * HID; idx += NTHR) {
        int g   = idx / (16 * HID);
        int rem = idx - g * (16 * HID);
        int r   = rem / HID;            // col within slice 0..15
        int k   = rem - r * HID;
        wlds[idx] = (bf16_t)w_hh[(size_t)(g * HID + j + r) * HID + k];
    }
    __syncthreads();

    // ---- One-time: gi = x @ w_ih.T + b_ih for this wave's 3 gate tiles ----
    // Register layout matches WMMA C layout: slot v -> row rowbase+v+8*hi, col j+ln
    const int c0 = 0 * HID + j + ln;
    const int c1 = 1 * HID + j + ln;
    const int c2 = 2 * HID + j + ln;
    float gi0[8], gi1[8], gi2[8];
    #pragma unroll
    for (int v = 0; v < 8; ++v) { gi0[v] = 0.f; gi1[v] = 0.f; gi2[v] = 0.f; }
    for (int k = 0; k < IN_DIM; ++k) {
        float w0 = w_ih[(size_t)c0 * IN_DIM + k];
        float w1 = w_ih[(size_t)c1 * IN_DIM + k];
        float w2 = w_ih[(size_t)c2 * IN_DIM + k];
        #pragma unroll
        for (int v = 0; v < 8; ++v) {
            float xv = x[(size_t)(rowbase + v + 8 * hi) * IN_DIM + k];
            gi0[v] += xv * w0;
            gi1[v] += xv * w1;
            gi2[v] += xv * w2;
        }
    }
    {
        float bi0 = b_ih[c0], bi1 = b_ih[c1], bi2 = b_ih[c2];
        #pragma unroll
        for (int v = 0; v < 8; ++v) { gi0[v] += bi0; gi1[v] += bi1; gi2[v] += bi2; }
    }
    const float bh0 = b_hh[c0], bh1 = b_hh[c1], bh2 = b_hh[c2];

    // ---- h0 = 0: init buffer 0 (each wave owns its tile, no overlap) ----
    #pragma unroll
    for (int v = 0; v < 8; ++v) {
        int m = rowbase + v + 8 * hi;
        hbuf[(size_t)m * HID + j + ln] = (bf16_t)0.0f;
    }
    float hprev[8];
    #pragma unroll
    for (int v = 0; v < 8; ++v) hprev[v] = 0.0f;

    unsigned bar = 1;
    grid_barrier(counter, bar * NWG); ++bar;   // h0 visible everywhere

    // ---- 1024 dependent steps ----
    for (int s = 0; s < SEQ; ++s) {
        const bf16_t* __restrict__ hcur = hbuf + (size_t)(s & 1) * (BATCH * HID);
        bf16_t* __restrict__       hnxt = hbuf + (size_t)((s + 1) & 1) * (BATCH * HID);

        v8f acc0 = {}, acc1 = {}, acc2 = {};

        // A fragment row for this lane: M = ln (ISA 16-bit A layout)
        const bf16_t* __restrict__ arow = hcur + (size_t)(rowbase + ln) * HID;

        #pragma unroll 4
        for (int ko = 0; ko < HID; ko += 32) {
            // A: lanes 0-15 K 0-15; lanes 16-31 K 16-31
            int abase = ko + hi * 16;
            v8bf alo = *(const v8bf*)(arow + abase);
            v8bf ah2 = *(const v8bf*)(arow + abase + 8);
            v16bf a;
            #pragma unroll
            for (int e = 0; e < 8; ++e) { a[e] = alo[e]; a[e + 8] = ah2[e]; }

            // B: lane = N; lanes 0-15 K ko..ko+15, lanes 16-31 K ko+16..ko+31
            int bbase = ko + hi * 16;
            v16bf b0 = *(const v16bf*)(&wlds[(0 * 16 + ln) * HID + bbase]);
            v16bf b1 = *(const v16bf*)(&wlds[(1 * 16 + ln) * HID + bbase]);
            v16bf b2 = *(const v16bf*)(&wlds[(2 * 16 + ln) * HID + bbase]);

            acc0 = __builtin_amdgcn_wmma_f32_16x16x32_bf16(false, a, false, b0,
                                                           (short)0, acc0, false, false);
            acc1 = __builtin_amdgcn_wmma_f32_16x16x32_bf16(false, a, false, b1,
                                                           (short)0, acc1, false, false);
            acc2 = __builtin_amdgcn_wmma_f32_16x16x32_bf16(false, a, false, b2,
                                                           (short)0, acc2, false, false);
        }

        // ---- gating: all wave-local, fp32 ----
        #pragma unroll
        for (int v = 0; v < 8; ++v) {
            float r  = fast_sigmoid(gi0[v] + acc0[v] + bh0);
            float z  = fast_sigmoid(gi1[v] + acc1[v] + bh1);
            float n  = fast_tanh(gi2[v] + r * (acc2[v] + bh2));
            float hn = (1.0f - z) * n + z * hprev[v];
            hprev[v] = hn;

            int m = rowbase + v + 8 * hi;
            out[((size_t)m * SEQ + s) * HID + (j + ln)] = hn;           // fp32 output
            hnxt[(size_t)m * HID + (j + ln)] = (bf16_t)hn;              // bf16 for next matmul
        }

        grid_barrier(counter, bar * NWG); ++bar;  // publish h_{s+1} before next step
    }
}

extern "C" void kernel_launch(void* const* d_in, const int* in_sizes, int n_in,
                              void* d_out, int out_size, void* d_ws, size_t ws_size,
                              hipStream_t stream) {
    (void)in_sizes; (void)n_in; (void)out_size; (void)ws_size;
    const float* x    = (const float*)d_in[0];
    const float* w_ih = (const float*)d_in[1];
    const float* w_hh = (const float*)d_in[2];
    const float* b_ih = (const float*)d_in[3];
    const float* b_hh = (const float*)d_in[4];
    float* out = (float*)d_out;

    // d_ws layout: [0,256): barrier counter (+pad); [256, 256+256KiB): double-buffered h (bf16)
    unsigned* counter = (unsigned*)d_ws;
    bf16_t*   hbuf    = (bf16_t*)((char*)d_ws + 256);

    hipMemsetAsync(d_ws, 0, 256, stream);  // reset barrier counter (capture-safe)

    gru_decoder_persistent<<<NWG, NTHR, 0, stream>>>(
        x, w_ih, w_hh, b_ih, b_hh, out, counter, hbuf);
}
